// MultiHeadSelfAttention_82016695484980
// MI455X (gfx1250) — matrix operand order repaired and checked by reading, not compiled, not run on hardware
//
#include <hip/hip_runtime.h>
#include <hip/hip_bf16.h>

// ---------------------------------------------------------------------------
// MHA forward for B=2,S=2048,D=1024,H=16,HD=64 on gfx1250 (wave32, WMMA).
// d_out = [ctx (B*S*D fp32) | attn (B*H*S*S fp32)].
// Roofline: the 512MB attn store dominates -> write it exactly once with
// non-temporal b128 stores (keep L2 for Q/K/V reuse); all GEMMs run on
// v_wmma_f32_16x16x32_bf16 (bf16 operands, fp32 accumulate).
// ---------------------------------------------------------------------------

typedef __attribute__((ext_vector_type(16))) __bf16 v16bf;
typedef __attribute__((ext_vector_type(8)))  __bf16 v8bf;
typedef __attribute__((ext_vector_type(8)))  float  v8f;
typedef __attribute__((ext_vector_type(4)))  float  v4f;

#define NB   2
#define NS   2048
#define ND   1024
#define NH   16
#define NHD  64
#define NM   (NB*NS)            // 4096 rows of X
#define CTX_ELEMS (NB*NS*ND)    // 4,194,304
#define SROW 2052               // padded LDS score-row stride (banks shift by 4)

// Workspace layout (bf16 elements)
#define XB_OFF 0u                                   // [4096][1024]
#define WT_OFF (XB_OFF + (unsigned)(NM*ND))         // 3 x [1024][1024] (transposed)
#define QB_OFF (WT_OFF + 3u*ND*ND)                  // [B][H][S][HD], pre-scaled 1/8
#define KB_OFF (QB_OFF + (unsigned)(NB*NH*NS*NHD))  // [B][H][S][HD]
#define VT_OFF (KB_OFF + (unsigned)(NB*NH*NS*NHD))  // [B][H][HD][S]

__device__ __forceinline__ v8f wmma_bf16(v16bf a, v16bf b, v8f c) {
  return __builtin_amdgcn_wmma_f32_16x16x32_bf16(false, a, false, b, (short)0, c,
                                                 false, false);
}

// ---- 1) fp32 -> bf16 convert of hidden_states -----------------------------
__global__ void cvt_x(const float* __restrict__ X, __bf16* __restrict__ Xb) {
  int i = blockIdx.x * blockDim.x + threadIdx.x;
  Xb[i] = (__bf16)X[i];
}

// ---- 2) transpose+convert W (so B-matrix reads are contraction-contiguous)
__global__ void cvt_wt(const float* __restrict__ Wq, const float* __restrict__ Wk,
                       const float* __restrict__ Wv, __bf16* __restrict__ Wt) {
  int i = blockIdx.x * blockDim.x + threadIdx.x;  // i = j*1024 + d
  int j = i >> 10, d = i & (ND - 1);
  const float* src = (blockIdx.y == 0) ? Wq : (blockIdx.y == 1) ? Wk : Wv;
  Wt[blockIdx.y * (ND * ND) + i] = (__bf16)src[d * ND + j];
}

// ---- 3) QKV projection: X[4096,1024] @ W[1024,1024] + bias ---------------
// grid (M/16=256, 8, 3), block 256 = 8 waves; each wave owns one 16x16 tile.
__global__ __launch_bounds__(256) void qkv_gemm(
    const __bf16* __restrict__ Xb, const __bf16* __restrict__ WtAll,
    const float* __restrict__ bq, const float* __restrict__ bk,
    const float* __restrict__ bv, __bf16* __restrict__ Qb,
    __bf16* __restrict__ Kb, __bf16* __restrict__ Vt) {
  const int wv = threadIdx.x >> 5, lane = threadIdx.x & 31;
  const int r = lane & 15, hi = lane >> 4;
  const int tileM = blockIdx.x;
  const int colTile = blockIdx.y * 8 + wv;
  const int z = blockIdx.z;                       // 0=Q 1=K 2=V
  const __bf16* Wt = WtAll + (size_t)z * (ND * ND);
  const float* bias = (z == 0) ? bq : (z == 1) ? bk : bv;

  const int arow = tileM * 16 + r;    // A: output row (lane%16)
  const int bcol = colTile * 16 + r;  // B: output col (lane%16)
  v8f acc = {};
#pragma unroll 4
  for (int kc = 0; kc < ND / 32; ++kc) {
    const int dbase = kc * 32;
    // A 16x32 bf16: lane holds K = [hi*16, hi*16+16) contiguous
    v8bf a0 = *(const v8bf*)(Xb + (size_t)arow * ND + dbase + hi * 16);
    v8bf a1 = *(const v8bf*)(Xb + (size_t)arow * ND + dbase + hi * 16 + 8);
    v16bf a;
#pragma unroll
    for (int t = 0; t < 8; ++t) { a[t] = a0[t]; a[t + 8] = a1[t]; }
    // B 32x16 bf16: lane holds 16 contiguous contraction values at hi*16
    v16bf bm = *(const v16bf*)(Wt + (size_t)bcol * ND + dbase + hi * 16);
    acc = wmma_bf16(a, bm, acc);
  }
  const int j = colTile * 16 + r;
  const float bj = bias[j];
  const int h = j >> 6, hd = j & 63;
#pragma unroll
  for (int i = 0; i < 8; ++i) {   // C/D: element (m = i + 8*hi, n = lane%16)
    const int m = tileM * 16 + i + 8 * hi;
    const int b = m >> 11, s = m & (NS - 1);
    const float val = acc[i] + bj;
    if (z == 0)
      Qb[(((size_t)(b * NH + h) * NS) + s) * NHD + hd] = (__bf16)(val * 0.125f);
    else if (z == 1)
      Kb[(((size_t)(b * NH + h) * NS) + s) * NHD + hd] = (__bf16)val;
    else
      Vt[(((size_t)(b * NH + h) * NHD) + hd) * NS + s] = (__bf16)val;
  }
}

// ---- 4) fused scores -> softmax -> attn store -> attn@V ------------------
// grid (S/16=128, B*H=32), block 256 (8 waves).
// LDS: scores 16x2052 fp32 (~128KB, padded) + partials 8x16x64 fp32 (32KB).
__global__ __launch_bounds__(256) void attn_kernel(
    const __bf16* __restrict__ Qb, const __bf16* __restrict__ Kb,
    const __bf16* __restrict__ Vt, const long long* __restrict__ mask,
    float* __restrict__ ctx_out, float* __restrict__ attn_out) {
  extern __shared__ float smem[];
  float* S_ = smem;                // [16][SROW] scores / probs
  float* R_ = smem + 16 * SROW;    // [8][16][64] partials; softmax scratch

  const int tid = threadIdx.x;
  const int wv = tid >> 5, lane = tid & 31, r = lane & 15, hi = lane >> 4;
  const int bh = blockIdx.y, b = bh >> 4, h = bh & 15;
  const int qb = blockIdx.x * 16;
  const __bf16* Qh = Qb + (size_t)bh * NS * NHD;
  const __bf16* Kh = Kb + (size_t)bh * NS * NHD;
  const __bf16* Vh = Vt + (size_t)bh * NHD * NS;

  // Q A-tiles for both HD halves, held in registers for the whole block.
  v16bf aq0, aq1;
  {
    const __bf16* qrow = Qh + (size_t)(qb + r) * NHD;
    v8bf t0 = *(const v8bf*)(qrow + hi * 16);
    v8bf t1 = *(const v8bf*)(qrow + hi * 16 + 8);
    v8bf t2 = *(const v8bf*)(qrow + 32 + hi * 16);
    v8bf t3 = *(const v8bf*)(qrow + 40 + hi * 16);
#pragma unroll
    for (int t = 0; t < 8; ++t) {
      aq0[t] = t0[t]; aq0[t + 8] = t1[t];
      aq1[t] = t2[t]; aq1[t + 8] = t3[t];
    }
  }

  // ---- scores: each wave produces 16 key-tiles of 16 (full S = 2048) ----
  for (int nt = wv; nt < NS / 16; nt += 8) {
    const int kb0 = nt * 16;
    const __bf16* krow = Kh + (size_t)(kb0 + r) * NHD;
    v16bf b0 = *(const v16bf*)(krow + hi * 16);
    v16bf b1 = *(const v16bf*)(krow + 32 + hi * 16);
    v8f c = {};
    c = wmma_bf16(aq0, b0, c);
    c = wmma_bf16(aq1, b1, c);
    // additive mask bias; column (= key) is lane%16, same for all 8 elems
    const float mb = mask[b * NS + kb0 + r] ? 0.0f : -3.0e38f;
#pragma unroll
    for (int i = 0; i < 8; ++i) S_[(i + 8 * hi) * SROW + kb0 + r] = c[i] + mb;
  }
  __syncthreads();

  // ---- softmax: 16 threads per row, float4-vectorized over 2048 cols -----
  const int rr = tid >> 4, cc = tid & 15;
  const v4f* S4c = (const v4f*)S_;
  v4f* S4 = (v4f*)S_;
  const int rb4 = rr * (SROW / 4);        // SROW/4 = 513 float4 per row
  float lmax = -3.4e38f;
  for (int i = 0; i < 32; ++i) {
    v4f v = S4c[rb4 + cc + i * 16];
    lmax = fmaxf(fmaxf(fmaxf(fmaxf(lmax, v[0]), v[1]), v[2]), v[3]);
  }
  R_[tid] = lmax;
  __syncthreads();
  float rmax = -3.4e38f;
#pragma unroll
  for (int j2 = 0; j2 < 16; ++j2) rmax = fmaxf(rmax, R_[rr * 16 + j2]);
  __syncthreads();
  float lsum = 0.0f;
  for (int i = 0; i < 32; ++i) {
    v4f v = S4[rb4 + cc + i * 16];
    v4f e;
#pragma unroll
    for (int t = 0; t < 4; ++t) e[t] = __expf(v[t] - rmax);
    S4[rb4 + cc + i * 16] = e;
    lsum += e[0] + e[1] + e[2] + e[3];
  }
  R_[tid] = lsum;
  __syncthreads();
  float rsum = 0.0f;
#pragma unroll
  for (int j2 = 0; j2 < 16; ++j2) rsum += R_[rr * 16 + j2];
  const float inv = 1.0f / rsum;
  // single HBM pass over attn: non-temporal b128 stores (don't thrash L2)
  v4f* arow4 = (v4f*)(attn_out + ((size_t)bh * NS + qb + rr) * NS);
  for (int i = 0; i < 32; ++i) {
    v4f v = S4[rb4 + cc + i * 16];
#pragma unroll
    for (int t = 0; t < 4; ++t) v[t] *= inv;
    S4[rb4 + cc + i * 16] = v;            // keep normalized probs for attn@V
    __builtin_nontemporal_store(v, arow4 + cc + i * 16);
  }
  __syncthreads();

  // ---- ctx = probs[16,2048] @ V[2048,64]; wave owns 256-key slice --------
  v8f acc0 = {}, acc1 = {}, acc2 = {}, acc3 = {};
  for (int kc = 0; kc < 8; ++kc) {
    const int kbase = wv * 256 + kc * 32;
    const float* sp = S_ + r * SROW + kbase;
    v4f a0 = *(const v4f*)(sp + hi * 16);
    v4f a1 = *(const v4f*)(sp + hi * 16 + 4);
    v4f a2 = *(const v4f*)(sp + hi * 16 + 8);
    v4f a3 = *(const v4f*)(sp + hi * 16 + 12);
    v16bf a;
#pragma unroll
    for (int t = 0; t < 4; ++t) {
      a[t]      = (__bf16)a0[t];
      a[t + 4]  = (__bf16)a1[t];
      a[t + 8]  = (__bf16)a2[t];
      a[t + 12] = (__bf16)a3[t];
    }
    v16bf b0 = *(const v16bf*)(Vh + (size_t)(0 * 16 + r) * NS + kbase + hi * 16);
    v16bf b1 = *(const v16bf*)(Vh + (size_t)(1 * 16 + r) * NS + kbase + hi * 16);
    v16bf b2 = *(const v16bf*)(Vh + (size_t)(2 * 16 + r) * NS + kbase + hi * 16);
    v16bf b3 = *(const v16bf*)(Vh + (size_t)(3 * 16 + r) * NS + kbase + hi * 16);
    acc0 = wmma_bf16(a, b0, acc0);
    acc1 = wmma_bf16(a, b1, acc1);
    acc2 = wmma_bf16(a, b2, acc2);
    acc3 = wmma_bf16(a, b3, acc3);
  }
  float* Rw = R_ + wv * (16 * 64);
#pragma unroll
  for (int i = 0; i < 8; ++i) {
    const int m = i + 8 * hi;
    Rw[m * 64 +  0 + r] = acc0[i];
    Rw[m * 64 + 16 + r] = acc1[i];
    Rw[m * 64 + 32 + r] = acc2[i];
    Rw[m * 64 + 48 + r] = acc3[i];
  }
  __syncthreads();

  // ---- cross-wave reduction + fp32 ctx store -----------------------------
  for (int e = tid; e < 16 * 64; e += 256) {
    const int m = e >> 6, c2 = e & 63;
    float s = 0.0f;
#pragma unroll
    for (int w = 0; w < 8; ++w) s += R_[w * (16 * 64) + e];
    ctx_out[((size_t)(b * NS + qb + m)) * ND + h * NHD + c2] = s;
  }
}

// ---------------------------------------------------------------------------
extern "C" void kernel_launch(void* const* d_in, const int* in_sizes, int n_in,
                              void* d_out, int out_size, void* d_ws, size_t ws_size,
                              hipStream_t stream) {
  (void)in_sizes; (void)n_in; (void)out_size; (void)ws_size;
  const float*     X    = (const float*)d_in[0];
  const long long* mask = (const long long*)d_in[1];
  const float* Wq = (const float*)d_in[2];
  const float* bq = (const float*)d_in[3];
  const float* Wk = (const float*)d_in[4];
  const float* bk = (const float*)d_in[5];
  const float* Wv = (const float*)d_in[6];
  const float* bv = (const float*)d_in[7];

  __bf16* ws = (__bf16*)d_ws;
  __bf16* Xb = ws + XB_OFF;
  __bf16* Wt = ws + WT_OFF;
  __bf16* Qb = ws + QB_OFF;
  __bf16* Kb = ws + KB_OFF;
  __bf16* Vt = ws + VT_OFF;

  float* ctx  = (float*)d_out;
  float* attn = (float*)d_out + CTX_ELEMS;

  cvt_x<<<(NM * ND) / 256, 256, 0, stream>>>(X, Xb);
  cvt_wt<<<dim3((ND * ND) / 256, 3), 256, 0, stream>>>(Wq, Wk, Wv, Wt);
  qkv_gemm<<<dim3(NM / 16, (ND / 16) / 8, 3), 256, 0, stream>>>(
      Xb, Wt, bq, bk, bv, Qb, Kb, Vt);
  const size_t smem = (16 * SROW + 8 * 16 * 64) * sizeof(float);  // ~160 KB
  attn_kernel<<<dim3(NS / 16, NB * NH), 256, smem, stream>>>(
      Qb, Kb, Vt, mask, ctx, attn);
}
